// NormalizingFlowPolicy_70574902608353
// MI455X (gfx1250) — matrix operand order repaired and checked by reading, not compiled, not run on hardware
//
#include <hip/hip_runtime.h>
#include <hip/hip_bf16.h>

// MI455X / gfx1250 implementation of the normalizing-flow policy kernel.
// 2 waves (64 threads) = one batch element; each wave owns a 32-column block
// of the 64x64 Jacobian tangent (all inter-matmul scalings are row-diagonal,
// so column blocks are independent). Jacobian forward-mode propagation runs
// on v_wmma_f32_16x16x32_f16 (f16 operands, f32 accumulate).
//
// All diagonal factors are folded into the A-operands at f16-conversion time:
//   diag(c)*W2*D1*W1*D0*W0*X = (diag(c)*W2*diag(d1)) * ((W1*diag(d0)) * (W0*X))
// so the chain is pure WMMA -> cvt -> LDS -> WMMA with no output scaling on
// the critical path (kills the WMMA->VALU RAW hazard v_nops). The coupling
// combine uses elementwise pre-scale by exp(s) + WMMA C-accumulate.
// 64x64 solve in f32 LDS (no pivoting: J ~ I + O(0.1)).

typedef __attribute__((ext_vector_type(16))) _Float16 v16h;
typedef __attribute__((ext_vector_type(8)))  _Float16 v8h;
typedef __attribute__((ext_vector_type(8)))  float    v8f;

// Compiler-only ordering fence for wave-private LDS producer/consumer:
// same-wave DS ops execute in issue order in hardware, we only need to stop
// the compiler from reordering the memory ops.
__device__ __forceinline__ void lds_fence() { asm volatile("" ::: "memory"); }

// ---- A-operand loader: f32 row-major weight [M][K] -> 16x32 f16 A layout ----
// ISA layout: lanes 0-15 (M=lane): halves 0..7 = K0..7, halves 8..15 = K16..23
//             lanes 16-31 (M=lane-16): halves 0..7 = K8..15, 8..15 = K24..31
// Optional per-row and per-column scales (LDS) folded into the conversion.
// Column scale is only meaningful for k < Kv (elements are 0 beyond), so the
// index is clamped with &7 to stay in bounds of the 8-entry diagonal.
__device__ __forceinline__ v16h load_A_tile(const float* base, int rs, int Mv,
                                            int Kv, int lane,
                                            const float* rscale, int rofs,
                                            const float* cscale) {
  int M  = lane & 15;
  int kb = (lane < 16) ? 0 : 8;
  float sr = rscale ? rscale[rofs + M] : 1.0f;
  v16h a;
#pragma unroll
  for (int i = 0; i < 8; ++i) {
    int k1 = kb + i;
    int k2 = kb + 16 + i;
    float c1 = cscale ? cscale[k1 & 7] : 1.0f;
    float c2 = cscale ? cscale[k2 & 7] : 1.0f;
    float f1 = (M < Mv && k1 < Kv) ? base[M * rs + k1] * sr * c1 : 0.0f;
    float f2 = (M < Mv && k2 < Kv) ? base[M * rs + k2] * sr * c2 : 0.0f;
    a[i]     = (_Float16)f1;
    a[8 + i] = (_Float16)f2;
  }
  return a;
}

// ---- C-tile (f32) -> LDS staging buffer, f16, [N][K] with K-stride 32 ----
// C layout: VGPR r of lane l holds row M = (l/16)*8 + r (+16*rt), col N = nt*16 + l%16
__device__ __forceinline__ void stage_write(_Float16* stage, const v8f& t,
                                            int rt, int nt, int lane) {
  int N  = nt * 16 + (lane & 15);
  int K0 = rt * 16 + ((lane >> 4) << 3);
  v8h h;
#pragma unroll
  for (int i = 0; i < 8; ++i) h[i] = (_Float16)t[i];
  *reinterpret_cast<v8h*>(&stage[N * 32 + K0]) = h;
}

// ---- B-operand (32x16 f16): lanes 0-15 = K0..7,K16..23; lanes 16-31 = K8..15,K24..31 ----
__device__ __forceinline__ v16h stage_read_full(const _Float16* stage, int nt,
                                                int lane) {
  int N  = nt * 16 + (lane & 15);
  int kb = (lane < 16) ? 0 : 8;
  v8h p0 = *reinterpret_cast<const v8h*>(&stage[N * 32 + kb]);
  v8h p1 = *reinterpret_cast<const v8h*>(&stage[N * 32 + kb + 16]);
  v16h b;
#pragma unroll
  for (int i = 0; i < 8; ++i) { b[i] = p0[i]; b[8 + i] = p1[i]; }
  return b;
}

// ---- forward-mode tangent through one MLP, accumulated into `out`:
//      out += diag(w2scale) * W2 * D1 * W1 * D0 * W0 * X
// The wave's two column tiles (nt = wv*2, wv*2+1) are processed interleaved.
// stageX: full-K tangent staging. stageH: hidden staging, K>=16 stays zero.
__device__ __forceinline__ void tangent_chain(
    const float* W0b, const float* W1b, const float* W2b,
    const float* w2scale,  // nullptr or LDS[32] row scale folded into W2
    const v8f X[2][2], v8f out[2][2],
    _Float16* stageX, _Float16* stageH, const float* d0buf,
    const float* d1buf, int lane, int wv) {
  const v8f vz = {0.f, 0.f, 0.f, 0.f, 0.f, 0.f, 0.f, 0.f};
  const int nt0 = wv * 2;
  const int nt1 = wv * 2 + 1;

  v16h A0  = load_A_tile(W0b, 32, 8, 32, lane, nullptr, 0, nullptr);
  v16h A1  = load_A_tile(W1b, 8, 8, 8, lane, nullptr, 0, d0buf);
  v16h A2a = load_A_tile(W2b, 8, 16, 8, lane, w2scale, 0, d1buf);
  v16h A2b = load_A_tile(W2b + 16 * 8, 8, 16, 8, lane, w2scale, 16, d1buf);

  lds_fence();
  stage_write(stageX, X[0][0], 0, nt0, lane);
  stage_write(stageX, X[1][0], 1, nt0, lane);
  stage_write(stageX, X[0][1], 0, nt1, lane);
  stage_write(stageX, X[1][1], 1, nt1, lane);
  lds_fence();

  // layer 0: (8x32 padded to 16x32) @ (32x16) x2 tiles
  v16h B0 = stage_read_full(stageX, nt0, lane);
  v16h B1 = stage_read_full(stageX, nt1, lane);
  v8f h0 = __builtin_amdgcn_wmma_f32_16x16x32_f16(false, A0, false, B0,
                                                  (short)0, vz, false, false);
  v8f h1 = __builtin_amdgcn_wmma_f32_16x16x32_f16(false, A0, false, B1,
                                                  (short)0, vz, false, false);
  lds_fence();
  stage_write(stageH, h0, 0, nt0, lane);
  stage_write(stageH, h1, 0, nt1, lane);
  lds_fence();

  // layer 1: (W1*diag(d0), 8x8 padded) @ (8x16) x2 tiles
  B0 = stage_read_full(stageH, nt0, lane);
  B1 = stage_read_full(stageH, nt1, lane);
  h0 = __builtin_amdgcn_wmma_f32_16x16x32_f16(false, A1, false, B0, (short)0,
                                              vz, false, false);
  h1 = __builtin_amdgcn_wmma_f32_16x16x32_f16(false, A1, false, B1, (short)0,
                                              vz, false, false);
  lds_fence();
  stage_write(stageH, h0, 0, nt0, lane);
  stage_write(stageH, h1, 0, nt1, lane);
  lds_fence();

  // layer 2: (diag(c)*W2*diag(d1), two 16x8 tiles) @ (8x16) x2, C-accumulate
  B0 = stage_read_full(stageH, nt0, lane);
  B1 = stage_read_full(stageH, nt1, lane);
  out[0][0] = __builtin_amdgcn_wmma_f32_16x16x32_f16(
      false, A2a, false, B0, (short)0, out[0][0], false, false);
  out[1][0] = __builtin_amdgcn_wmma_f32_16x16x32_f16(
      false, A2b, false, B0, (short)0, out[1][0], false, false);
  out[0][1] = __builtin_amdgcn_wmma_f32_16x16x32_f16(
      false, A2a, false, B1, (short)0, out[0][1], false, false);
  out[1][1] = __builtin_amdgcn_wmma_f32_16x16x32_f16(
      false, A2b, false, B1, (short)0, out[1][1], false, false);
}

// ---- f32 scalar forward through one MLP (tid<8 lanes write shared state,
//      every thread returns output row (tid&31)) ----
__device__ __forceinline__ float forward_subnet(
    const float* W0b, const float* b0b, const float* W1b, const float* b1b,
    const float* W2b, const float* b2b, const float* inbuf, float* hbuf,
    float* h2buf, float* d0buf, float* d1buf, int tid) {
  if (tid < 8) {
    float acc = b0b[tid];
#pragma unroll
    for (int k = 0; k < 32; ++k) acc += W0b[tid * 32 + k] * inbuf[k];
    float h = tanhf(acc);
    hbuf[tid]  = h;
    d0buf[tid] = 1.0f - h * h;
  }
  lds_fence();  // hbuf producer/consumer are both wave-0 lanes
  if (tid < 8) {
    float acc = b1b[tid];
#pragma unroll
    for (int k = 0; k < 8; ++k) acc += W1b[tid * 8 + k] * hbuf[k];
    float h = tanhf(acc);
    h2buf[tid] = h;
    d1buf[tid] = 1.0f - h * h;
  }
  __syncthreads();  // h2buf/d0buf/d1buf consumed by both waves
  float o = b2b[tid & 31];
#pragma unroll
  for (int k = 0; k < 8; ++k) o += W2b[(tid & 31) * 8 + k] * h2buf[k];
  return o;
}

__global__ __launch_bounds__(64) void NormalizingFlowPolicy_70574902608353_kernel(
    const float* __restrict__ x, const float* __restrict__ xs,
    const float* __restrict__ W0g, const float* __restrict__ b0g,
    const float* __restrict__ W1g, const float* __restrict__ b1g,
    const float* __restrict__ W2g, const float* __restrict__ b2g,
    float* __restrict__ out) {
  __shared__ alignas(16) _Float16 stageX[64 * 32];  // tangent staging, 4KB
  __shared__ alignas(16) _Float16 stageH[64 * 32];  // hidden staging, 4KB
  __shared__ float Amat[64 * 65];                   // solve workspace, padded
  __shared__ float inbuf[32], hbuf[8], h2buf[8], d0buf[16], d1buf[16];
  __shared__ float abuf[32], cbuf[32], gbuf[64], ybuf[64], redbuf[2];

  const int tid   = threadIdx.x;
  const int lane  = tid & 31;
  const int wv    = tid >> 5;  // wave 0: columns 0..31, wave 1: columns 32..63
  const int b     = blockIdx.x;
  const int mbase = (lane >> 4) << 3;

  if (tid < 8) { d0buf[8 + tid] = 0.0f; d1buf[8 + tid] = 0.0f; }
  {  // zero the K=16..31 half of stageH once; it is never written again
    v8h hz;
#pragma unroll
    for (int i = 0; i < 8; ++i) hz[i] = (_Float16)0.0f;
    *reinterpret_cast<v8h*>(&stageH[tid * 32 + 16]) = hz;
    *reinterpret_cast<v8h*>(&stageH[tid * 32 + 24]) = hz;
  }

  float lo_r = x[b * 64 + lane];        // row `lane` of lo (both waves)
  float up_r = x[b * 64 + 32 + lane];   // row `lane` of up (both waves)

  // this wave's 32-column block of d(lo)/dv and d(up)/dv (32x32 f32 each)
  v8f lo_t[2][2], up_t[2][2];
#pragma unroll
  for (int rt = 0; rt < 2; ++rt)
#pragma unroll
    for (int c = 0; c < 2; ++c)
#pragma unroll
      for (int r = 0; r < 8; ++r) {
        int M = rt * 16 + mbase + r;
        int N = (wv * 2 + c) * 16 + (lane & 15);
        lo_t[rt][c][r] = (N == M) ? 1.0f : 0.0f;
        up_t[rt][c][r] = (N == 32 + M) ? 1.0f : 0.0f;
      }

  for (int f = 0; f < 8; ++f) {
    const int sb = f * 4;

    // ================= up update: up = t1(lo) + up * exp(s1(lo)) =========
    if (tid < 32) inbuf[tid] = lo_r;
    __syncthreads();
    float s1 = forward_subnet(W0g + (sb + 1) * 256, b0g + (sb + 1) * 8,
                              W1g + (sb + 1) * 64, b1g + (sb + 1) * 8,
                              W2g + (sb + 1) * 256, b2g + (sb + 1) * 32, inbuf,
                              hbuf, h2buf, d0buf, d1buf, tid);
    float a1 = __expf(s1);
    if (tid < 32) { abuf[tid] = a1; cbuf[tid] = up_r * a1; }
    __syncthreads();
    // up_t *= diag(exp(s1))
#pragma unroll
    for (int rt = 0; rt < 2; ++rt)
#pragma unroll
      for (int c = 0; c < 2; ++c)
#pragma unroll
        for (int r = 0; r < 8; ++r)
          up_t[rt][c][r] *= abuf[rt * 16 + mbase + r];
    // up_t += diag(up*exp(s1)) * s1_t   (diag(c) folded into W2 A-operand)
    tangent_chain(W0g + (sb + 1) * 256, W1g + (sb + 1) * 64,
                  W2g + (sb + 1) * 256, cbuf, lo_t, up_t, stageX, stageH,
                  d0buf, d1buf, lane, wv);
    __syncthreads();  // chain d-reads done before t-subnet overwrites them
    float t1 = forward_subnet(W0g + sb * 256, b0g + sb * 8, W1g + sb * 64,
                              b1g + sb * 8, W2g + sb * 256, b2g + sb * 32,
                              inbuf, hbuf, h2buf, d0buf, d1buf, tid);
    // up_t += t1_t
    tangent_chain(W0g + sb * 256, W1g + sb * 64, W2g + sb * 256, nullptr, lo_t,
                  up_t, stageX, stageH, d0buf, d1buf, lane, wv);
    up_r = t1 + up_r * a1;
    __syncthreads();

    // ================= lo update: lo = t2(up) + lo * exp(s2(up)) =========
    if (tid < 32) inbuf[tid] = up_r;
    __syncthreads();
    float s2 = forward_subnet(W0g + (sb + 3) * 256, b0g + (sb + 3) * 8,
                              W1g + (sb + 3) * 64, b1g + (sb + 3) * 8,
                              W2g + (sb + 3) * 256, b2g + (sb + 3) * 32, inbuf,
                              hbuf, h2buf, d0buf, d1buf, tid);
    float a2 = __expf(s2);
    if (tid < 32) { abuf[tid] = a2; cbuf[tid] = lo_r * a2; }
    __syncthreads();
#pragma unroll
    for (int rt = 0; rt < 2; ++rt)
#pragma unroll
      for (int c = 0; c < 2; ++c)
#pragma unroll
        for (int r = 0; r < 8; ++r)
          lo_t[rt][c][r] *= abuf[rt * 16 + mbase + r];
    tangent_chain(W0g + (sb + 3) * 256, W1g + (sb + 3) * 64,
                  W2g + (sb + 3) * 256, cbuf, up_t, lo_t, stageX, stageH,
                  d0buf, d1buf, lane, wv);
    __syncthreads();
    float t2 = forward_subnet(W0g + (sb + 2) * 256, b0g + (sb + 2) * 8,
                              W1g + (sb + 2) * 64, b1g + (sb + 2) * 8,
                              W2g + (sb + 2) * 256, b2g + (sb + 2) * 32, inbuf,
                              hbuf, h2buf, d0buf, d1buf, tid);
    tangent_chain(W0g + (sb + 2) * 256, W1g + (sb + 2) * 64,
                  W2g + (sb + 2) * 256, nullptr, up_t, lo_t, stageX, stageH,
                  d0buf, d1buf, lane, wv);
    lo_r = t2 + lo_r * a2;
    __syncthreads();
  }

  // ================= dump J to LDS and solve J y = -2 (x - x*) ===========
#pragma unroll
  for (int rt = 0; rt < 2; ++rt)
#pragma unroll
    for (int c = 0; c < 2; ++c)
#pragma unroll
      for (int r = 0; r < 8; ++r) {
        int M = rt * 16 + mbase + r;
        int N = (wv * 2 + c) * 16 + (lane & 15);
        Amat[M * 65 + N]        = lo_t[rt][c][r];
        Amat[(32 + M) * 65 + N] = up_t[rt][c][r];
      }
  gbuf[tid] = -2.0f * (x[b * 64 + tid] - xs[b * 64 + tid]);
  __syncthreads();

  // Gaussian elimination, no pivoting (J ~ I + O(0.1): diagonally dominant)
  for (int k = 0; k < 63; ++k) {
    float pivinv = 1.0f / Amat[k * 65 + k];
    int i = k + 1 + tid;  // one row per thread (64 threads, 64 rows)
    if (i < 64) {
      float m = Amat[i * 65 + k] * pivinv;
      for (int j = k + 1; j < 64; ++j) Amat[i * 65 + j] -= m * Amat[k * 65 + j];
      gbuf[i] -= m * gbuf[k];
    }
    __syncthreads();
  }
  // back substitution: per-wave shfl reduction, combine 2 waves through LDS
  for (int k = 63; k >= 0; --k) {
    float part = 0.0f;
    for (int j = k + 1 + tid; j < 64; j += 64) part += Amat[k * 65 + j] * ybuf[j];
#pragma unroll
    for (int off = 16; off > 0; off >>= 1) part += __shfl_down(part, off, 32);
    if (lane == 0) redbuf[wv] = part;
    __syncthreads();
    if (tid == 0) ybuf[k] = (gbuf[k] - redbuf[0] - redbuf[1]) / Amat[k * 65 + k];
    __syncthreads();
  }

  out[b * 64 + tid] = ybuf[tid];
}

extern "C" void kernel_launch(void* const* d_in, const int* in_sizes, int n_in,
                              void* d_out, int out_size, void* d_ws,
                              size_t ws_size, hipStream_t stream) {
  const float* x   = (const float*)d_in[0];
  const float* xs  = (const float*)d_in[1];
  const float* W0g = (const float*)d_in[2];
  const float* b0g = (const float*)d_in[3];
  const float* W1g = (const float*)d_in[4];
  const float* b1g = (const float*)d_in[5];
  const float* W2g = (const float*)d_in[6];
  const float* b2g = (const float*)d_in[7];
  float* outp      = (float*)d_out;

  const int Bn = in_sizes[0] / 64;  // 16384
  NormalizingFlowPolicy_70574902608353_kernel<<<Bn, 64, 0, stream>>>(
      x, xs, W0g, b0g, W1g, b1g, W2g, b2g, outp);
}
